// custom_71047349010451
// MI455X (gfx1250) — matrix operand order repaired and checked by reading, not compiled, not run on hardware
//
#include <hip/hip_runtime.h>
#include <math.h>

// ---------------------------------------------------------------------------
// ViT forward on gfx1250 (MI455X). All GEMMs via v_wmma_f32_16x16x32_f16.
// Memory-bound workload -> f16 activations/weights, fused epilogues,
// wide (b128) vector loads everywhere, LDS staging for strided accesses.
// ---------------------------------------------------------------------------

typedef __attribute__((ext_vector_type(16))) _Float16 v16h;
typedef __attribute__((ext_vector_type(8)))  _Float16 h8;    // 16B vector
typedef __attribute__((ext_vector_type(8)))  float    v8f;

#define BATCH 64
#define SEQ   197
#define HID   512
#define NL    20
#define NH    8
#define DH    64
#define MLPD  2048
#define OUTD  1000
#define INDIM 768
#define NPAT  196     // 14*14
#define MTOK  (BATCH*SEQ)    // 12608 = 64*197 -> 197 M-tiles of 64
#define MEMB  (BATCH*NPAT)   // 12544 = 64*196 -> 196 M-tiles of 64

__device__ __forceinline__ v8f wmma_f16(v16h a, v16h b, v8f c) {
  // D = A(16x32 f16) * B(32x16 f16) + C(16x16 f32)
  return __builtin_amdgcn_wmma_f32_16x16x32_f16(
      /*neg_a=*/false, a, /*neg_b=*/false, b,
      /*c_mod=*/(short)0, c, /*reuse_a=*/false, /*reuse_b=*/false);
}

// Assemble a 16x32 A-fragment from two contiguous 8-half runs.
// ISA layout: a[0..7] = row[base + half*8 .. +7], a[8..15] = row[base+16+half*8 .. +7]
__device__ __forceinline__ v16h make_frag(h8 lo, h8 hi) {
  v16h a;
#pragma unroll
  for (int i = 0; i < 8; ++i) { a[i] = lo[i]; a[8 + i] = hi[i]; }
  return a;
}

// ---------------------------------------------------------------------------
// Generic block-tiled WMMA GEMM: C[M,N] = A[M,K] (f16, row-major, lda) *
// B[K,N] (f16, row-major, ldb) + bias, fused epilogue.
// Block = 256 threads = 8 waves; tile 64x64; K step 32. M % 64 == 0 always.
// blockIdx.z batches (per-head QKV projections).
// ---------------------------------------------------------------------------
enum { EPI_F16 = 0, EPI_F16_GELU = 1, EPI_F32 = 2, EPI_F32_RES = 3, EPI_EMBED = 4 };

template <int EPI>
__global__ __launch_bounds__(256) void gemm_wmma_kernel(
    const _Float16* __restrict__ A, int lda, long aBS,
    const _Float16* __restrict__ B, int ldb, long bBS,
    const float* __restrict__ bias, long biasBS,
    float* Cf, int ldcf,
    _Float16* Ch, int ldch, long chBS,
    const float* __restrict__ pos,   // EPI_EMBED only
    int M, int N, int K) {
  // Row strides are multiples of 16B so b128 LDS ops stay aligned.
  __shared__ _Float16 As[64][48];   // 64x32 tile, stride 96B
  __shared__ _Float16 Bs[32][80];   // 32x64 tile, stride 160B

  const int z = blockIdx.z;
  A += (long)z * aBS;
  B += (long)z * bBS;
  bias += (long)z * biasBS;

  const int tid  = threadIdx.x;
  const int wave = tid >> 5;
  const int lane = tid & 31;
  const int half = lane >> 4;
  const int ln   = lane & 15;
  const int mrow = (wave >> 1) * 16;   // 0,16,32,48
  const int ncol = (wave & 1) * 32;    // 0,32
  const int m0   = blockIdx.y * 64;
  const int n0   = blockIdx.x * 64;

  const int ar = tid >> 2, ac = (tid & 3) * 8;   // A staging coords
  const int br = tid >> 3, bc = (tid & 7) * 8;   // B staging coords

  v8f acc[2] = {v8f{}, v8f{}};

  for (int k0 = 0; k0 < K; k0 += 32) {
    // Stage A tile 64x32: one b128 per thread.
    {
      const _Float16* src = A + (long)(m0 + ar) * lda + k0 + ac;
      *(h8*)&As[ar][ac] = *(const h8*)src;
      if (k0 + 32 < K) __builtin_prefetch(src + 32, 0, 3);  // global_prefetch
    }
    // Stage B tile 32x64: vector path unless the 8-run crosses N (head tail).
    {
      const _Float16* src = B + (long)(k0 + br) * ldb + n0 + bc;
      if (n0 + bc + 8 <= N) {
        *(h8*)&Bs[br][bc] = *(const h8*)src;
      } else {
#pragma unroll
        for (int j = 0; j < 8; ++j)
          Bs[br][bc + j] = (n0 + bc + j < N) ? src[j] : (_Float16)0.f;
      }
      if (k0 + 32 < K) __builtin_prefetch(src + (long)32 * ldb, 0, 3);
    }
    __syncthreads();

    // A fragment: two aligned 16B LDS reads.
    const _Float16* arow = &As[mrow + ln][half * 8];
    const v16h a = make_frag(*(const h8*)arow, *(const h8*)(arow + 16));

    // B fragments (32x16: element i -> K = half*8+i+(i&8), col = lane%16).
    v16h b0, b1;
#pragma unroll
    for (int i = 0; i < 16; ++i) {
      b0[i] = Bs[half * 8 + i + (i & 8)][ncol + ln];
      b1[i] = Bs[half * 8 + i + (i & 8)][ncol + 16 + ln];
    }
    acc[0] = wmma_f16(a, b0, acc[0]);
    acc[1] = wmma_f16(a, b1, acc[1]);
    __syncthreads();
  }

  // Epilogue. D layout: lane -> col = lane%16, row = half*8 + r.
#pragma unroll
  for (int t = 0; t < 2; ++t) {
    const int col = n0 + ncol + t * 16 + ln;
    const float bv = (col < N) ? bias[col] : 0.f;
#pragma unroll
    for (int r = 0; r < 8; ++r) {
      const int row = m0 + mrow + half * 8 + r;
      const float v = acc[t][r] + bv;
      if (col < N) {
        if constexpr (EPI == EPI_F16) {
          Ch[(long)z * chBS + (long)row * ldch + col] = (_Float16)v;
        } else if constexpr (EPI == EPI_F16_GELU) {
          const float g = 0.5f * v * (1.f + erff(v * 0.70710678118f));
          Ch[(long)z * chBS + (long)row * ldch + col] = (_Float16)g;
        } else if constexpr (EPI == EPI_F32) {
          Cf[(long)row * ldcf + col] = v;
        } else if constexpr (EPI == EPI_F32_RES) {
          float* p = Cf + (long)row * ldcf + col;
          *p = *p + v;
        } else {  // EPI_EMBED: row = b*196 + tok -> h[b, tok+1, col] + pos
          const int bimg = row / NPAT, tok = row % NPAT, s = tok + 1;
          Cf[((long)bimg * SEQ + s) * HID + col] = v + pos[(long)s * HID + col];
        }
      }
    }
  }
}

// ---------------------------------------------------------------------------
// Fused attention per (batch, head): S = QK^T -> masked softmax -> O = P V,
// accumulated directly into h (reference has no output projection).
// Block = 128 threads (4 waves); wave w owns query tiles {w, w+4, w+8, w+12}.
// Scores per wave live in an LDS strip Ps[16][224]; the P*V phase stages V
// chunks [32][64] into LDS block-uniformly (b128 loads), so the strided
// B-fragment gathers hit LDS instead of generating scalar global loads.
// ---------------------------------------------------------------------------
#define SEQP 224  // 197 padded to 7*32 for K=32 WMMA steps

__global__ __launch_bounds__(128) void attn_kernel(
    const _Float16* __restrict__ Q, const _Float16* __restrict__ Kk,
    const _Float16* __restrict__ V, float* __restrict__ H) {
  __shared__ float    Ps[4][16][SEQP];   // 57344 B
  __shared__ _Float16 Vt[32][72];        //  4608 B (stride 144B, 16B-mult)

  const int bh   = blockIdx.x;     // b*NH + head
  const int b    = bh >> 3;
  const int hd   = bh & 7;
  const int wave = threadIdx.x >> 5;
  const int lane = threadIdx.x & 31;
  const int half = lane >> 4;
  const int ln   = lane & 15;
  const float scale = 0.125f;      // 1/sqrt(64)

  // q/k/v layout: [B, SEQ, NH, DH] f16
  auto idx = [&](int s, int d) -> long {
    return (((long)b * SEQ + s) * NH + hd) * DH + d;
  };

  // V staging coords: 32x64 chunk, 16 halves per thread (two b128 loads).
  const int vr = threadIdx.x >> 2, vc = (threadIdx.x & 3) * 16;

  for (int it = 0; it < 4; ++it) {
    const int qt = wave + 4 * it;     // wave-uniform
    const bool active = (qt < 13);
    const int q0 = qt * 16;

    if (active) {
      // --- Q fragments: per K=32 step, two guarded b128 loads per lane ---
      v16h aq[2];
      const int qr = q0 + ln;
#pragma unroll
      for (int kc = 0; kc < 2; ++kc) {
        h8 qlo = {}, qhi = {};
        if (qr < SEQ) {
          const _Float16* qp = Q + idx(qr, kc * 32 + half * 8);
          qlo = *(const h8*)qp;
          qhi = *(const h8*)(qp + 16);
        }
        aq[kc] = make_frag(qlo, qhi);
      }

      // --- scores S[16, 208] = Q_tile * K^T, scaled, into LDS strip ---
      const int key = ln;  // + kt*16 below; B col = key index
      for (int kt = 0; kt < 13; ++kt) {
        v8f s = {};
        const int ky = kt * 16 + key;
#pragma unroll
        for (int kc = 0; kc < 2; ++kc) {
          // bk[i] = K[ky][kc*32 + half*8 + i + (i&8)]: two 8-half runs.
          h8 klo = {}, khi = {};
          if (ky < SEQ) {
            const _Float16* kp = Kk + idx(ky, kc * 32 + half * 8);
            klo = *(const h8*)kp;
            khi = *(const h8*)(kp + 16);
          }
          s = wmma_f16(aq[kc], make_frag(klo, khi), s);
        }
#pragma unroll
        for (int r = 0; r < 8; ++r)
          Ps[wave][half * 8 + r][kt * 16 + ln] = s[r] * scale;
      }
      // zero padded columns 208..223
      for (int i = lane; i < 16 * 16; i += 32)
        Ps[wave][i >> 4][208 + (i & 15)] = 0.f;
      asm volatile("s_wait_dscnt 0x0" ::: "memory");  // wave-local LDS visibility

      // --- masked softmax over keys [0,197), lanes 0..15 own one row ---
      if (lane < 16) {
        float mx = -3.0e38f;
        for (int c = 0; c < SEQ; ++c) mx = fmaxf(mx, Ps[wave][lane][c]);
        float sum = 0.f;
        for (int c = 0; c < SEQ; ++c) {
          const float e = __expf(Ps[wave][lane][c] - mx);
          Ps[wave][lane][c] = e;
          sum += e;
        }
        const float inv = 1.f / sum;
        for (int c = 0; c < SEQ; ++c) Ps[wave][lane][c] *= inv;
        for (int c = SEQ; c < 208; ++c) Ps[wave][lane][c] = 0.f;
      }
      asm volatile("s_wait_dscnt 0x0" ::: "memory");
    }

    // --- O[16,64] = P[16,224] * V[224,64]; V staged block-uniformly ---
    v8f o[4] = {v8f{}, v8f{}, v8f{}, v8f{}};
    for (int kc = 0; kc < 7; ++kc) {
      // Stage V chunk rows [kc*32, kc*32+32) cooperatively (b128 loads).
      {
        const int ky = kc * 32 + vr;
        h8 v0 = {}, v1 = {};
        if (ky < SEQ) {
          const _Float16* vp = V + idx(ky, vc);
          v0 = *(const h8*)vp;
          v1 = *(const h8*)(vp + 8);
        }
        *(h8*)&Vt[vr][vc]     = v0;
        *(h8*)&Vt[vr][vc + 8] = v1;
      }
      __syncthreads();
      if (active) {
        // P fragment: two aligned 32B f32 LDS reads, cvt to f16.
        const float* prow = &Ps[wave][ln][kc * 32 + half * 8];
        const v8f p0 = *(const v8f*)prow;
        const v8f p1 = *(const v8f*)(prow + 16);
        v16h ap;
#pragma unroll
        for (int i = 0; i < 8; ++i) {
          ap[i]     = (_Float16)p0[i];
          ap[8 + i] = (_Float16)p1[i];
        }
#pragma unroll
        for (int nt = 0; nt < 4; ++nt) {
          v16h bv;
#pragma unroll
          for (int i = 0; i < 16; ++i)
            bv[i] = Vt[half * 8 + i + (i & 8)][nt * 16 + ln];
          o[nt] = wmma_f16(ap, bv, o[nt]);
        }
      }
      __syncthreads();
    }

    if (active) {
#pragma unroll
      for (int nt = 0; nt < 4; ++nt) {
#pragma unroll
        for (int r = 0; r < 8; ++r) {
          const int q = q0 + half * 8 + r;
          if (q < SEQ) {
            float* p = H + ((long)b * SEQ + q) * HID + hd * DH + nt * 16 + ln;
            *p = *p + o[nt][r];
          }
        }
      }
    }
  }
}

// ---------------------------------------------------------------------------
// LayerNorm: one wave32 per token (512 elems -> 16/lane), shfl reductions.
// ---------------------------------------------------------------------------
__global__ __launch_bounds__(256) void ln_kernel(
    const float* __restrict__ X, const float* __restrict__ g,
    const float* __restrict__ bta, _Float16* __restrict__ Y, int ntok) {
  const int wave = threadIdx.x >> 5, lane = threadIdx.x & 31;
  const int tok = blockIdx.x * 8 + wave;
  if (tok >= ntok) return;
  const float* x = X + (long)tok * HID;
  float vals[16];
  float s = 0.f;
#pragma unroll
  for (int i = 0; i < 16; ++i) { vals[i] = x[lane + i * 32]; s += vals[i]; }
#pragma unroll
  for (int o = 16; o > 0; o >>= 1) s += __shfl_xor(s, o, 32);
  const float mean = s * (1.f / 512.f);
  float vs = 0.f;
#pragma unroll
  for (int i = 0; i < 16; ++i) { const float d = vals[i] - mean; vs += d * d; }
#pragma unroll
  for (int o = 16; o > 0; o >>= 1) vs += __shfl_xor(vs, o, 32);
  const float rstd = rsqrtf(vs * (1.f / 512.f) + 1e-5f);
  _Float16* y = Y + (long)tok * HID;
#pragma unroll
  for (int i = 0; i < 16; ++i) {
    const int c = lane + i * 32;
    y[c] = (_Float16)((vals[i] - mean) * rstd * g[c] + bta[c]);
  }
}

// ---------------------------------------------------------------------------
// Small helpers
// ---------------------------------------------------------------------------
__global__ void cvt_kernel(const float* __restrict__ s, _Float16* __restrict__ d, long n) {
  const long i = (long)blockIdx.x * blockDim.x + threadIdx.x;
  if (i < n) d[i] = (_Float16)s[i];
}

// qW[l][h] is [out=e][in=d]; WMMA B needs [k=d][n=e] -> 64x64 transpose.
__global__ void cvt_qkv_kernel(const float* __restrict__ s, _Float16* __restrict__ d, long nmat) {
  const long i = (long)blockIdx.x * blockDim.x + threadIdx.x;
  const long mat = i >> 12;
  if (mat >= nmat) return;
  const int r = (int)((i >> 6) & 63), c = (int)(i & 63);
  d[i] = (_Float16)s[(mat << 12) + ((long)c << 6) + r];
}

__global__ void pos_kernel(float* __restrict__ pos) {
  const int i = blockIdx.x;    // seq position
  const int j = threadIdx.x;   // hid dim (512)
  const float je = (float)(j - (j & 1));
  const float ang = (float)i * powf(10000.f, -je / 512.f);
  pos[(long)i * HID + j] = (j & 1) ? cosf(ang) : sinf(ang);
}

__global__ void patchify_kernel(const float* __restrict__ x, _Float16* __restrict__ P) {
  const long i = (long)blockIdx.x * blockDim.x + threadIdx.x;
  if (i >= (long)MEMB * INDIM) return;
  const int f = (int)(i % INDIM);
  const long rw = i / INDIM;
  const int tok = (int)(rw % NPAT);
  const int b = (int)(rw / NPAT);
  const int c = f >> 8, py = (f >> 4) & 15, px = f & 15;
  const int ph = tok / 14, pw = tok % 14;
  const int yy = ph * 16 + py, xx = pw * 16 + px;
  P[i] = (_Float16)x[(((long)b * 3 + c) * 224 + yy) * 224 + xx];
}

__global__ void clsrow_kernel(const float* __restrict__ cls,
                              const float* __restrict__ pos, float* __restrict__ H) {
  H[(long)blockIdx.x * SEQ * HID + threadIdx.x] = cls[threadIdx.x] + pos[threadIdx.x];
}

__global__ void extract_cls_kernel(const float* __restrict__ H, _Float16* __restrict__ C) {
  C[(long)blockIdx.x * HID + threadIdx.x] =
      (_Float16)H[(long)blockIdx.x * SEQ * HID + threadIdx.x];
}

__global__ __launch_bounds__(256) void softmax_kernel(
    const float* __restrict__ L, float* __restrict__ out) {
  __shared__ float rmax[8], rsum[8];
  const int b = blockIdx.x;
  const float* x = L + (long)b * OUTD;
  const int lane = threadIdx.x & 31, wv = threadIdx.x >> 5;
  float mx = -3.0e38f;
  for (int i = threadIdx.x; i < OUTD; i += 256) mx = fmaxf(mx, x[i]);
#pragma unroll
  for (int o = 16; o > 0; o >>= 1) mx = fmaxf(mx, __shfl_xor(mx, o, 32));
  if (lane == 0) rmax[wv] = mx;
  __syncthreads();
  mx = rmax[0];
#pragma unroll
  for (int w = 1; w < 8; ++w) mx = fmaxf(mx, rmax[w]);
  float s = 0.f;
  for (int i = threadIdx.x; i < OUTD; i += 256) s += __expf(x[i] - mx);
#pragma unroll
  for (int o = 16; o > 0; o >>= 1) s += __shfl_xor(s, o, 32);
  if (lane == 0) rsum[wv] = s;
  __syncthreads();
  s = 0.f;
  for (int w = 0; w < 8; ++w) s += rsum[w];
  const float inv = 1.f / s;
  for (int i = threadIdx.x; i < OUTD; i += 256)
    out[(long)b * OUTD + i] = __expf(x[i] - mx) * inv;
}

// ---------------------------------------------------------------------------
// Host driver. Workspace (~225 MB) carved from d_ws; everything rewritten
// every call (graph-replay deterministic).
// ---------------------------------------------------------------------------
extern "C" void kernel_launch(void* const* d_in, const int* in_sizes, int n_in,
                              void* d_out, int out_size, void* d_ws, size_t ws_size,
                              hipStream_t stream) {
  const float* x       = (const float*)d_in[0];
  const float* embed_W = (const float*)d_in[1];
  const float* embed_b = (const float*)d_in[2];
  const float* cls     = (const float*)d_in[3];
  const float* ln1_g   = (const float*)d_in[4];
  const float* ln1_b   = (const float*)d_in[5];
  const float* qW      = (const float*)d_in[6];
  const float* qb      = (const float*)d_in[7];
  const float* kW      = (const float*)d_in[8];
  const float* kb      = (const float*)d_in[9];
  const float* vW      = (const float*)d_in[10];
  const float* vb      = (const float*)d_in[11];
  const float* ln2_g   = (const float*)d_in[12];
  const float* ln2_b   = (const float*)d_in[13];
  const float* mlp1_W  = (const float*)d_in[14];
  const float* mlp1_b  = (const float*)d_in[15];
  const float* mlp2_W  = (const float*)d_in[16];
  const float* mlp2_b  = (const float*)d_in[17];
  const float* head_W  = (const float*)d_in[18];
  const float* head_b  = (const float*)d_in[19];

  char* ws = (char*)d_ws;
  size_t off = 0;
  auto carve = [&](size_t bytes) -> char* {
    char* p = ws + off;
    off = (off + bytes + 255) & ~(size_t)255;
    return p;
  };
  _Float16* wEmb = (_Float16*)carve((size_t)INDIM * HID * 2);
  _Float16* wQ   = (_Float16*)carve((size_t)NL * NH * DH * DH * 2);
  _Float16* wK   = (_Float16*)carve((size_t)NL * NH * DH * DH * 2);
  _Float16* wV   = (_Float16*)carve((size_t)NL * NH * DH * DH * 2);
  _Float16* wM1  = (_Float16*)carve((size_t)NL * HID * MLPD * 2);
  _Float16* wM2  = (_Float16*)carve((size_t)NL * MLPD * HID * 2);
  _Float16* wHd  = (_Float16*)carve((size_t)HID * OUTD * 2);
  float*    pos  = (float*)carve((size_t)SEQ * HID * 4);
  _Float16* Pm   = (_Float16*)carve((size_t)MEMB * INDIM * 2);
  float*    h    = (float*)carve((size_t)MTOK * HID * 4);
  _Float16* y16  = (_Float16*)carve((size_t)MTOK * HID * 2);
  _Float16* q16  = (_Float16*)carve((size_t)MTOK * HID * 2);
  _Float16* k16  = (_Float16*)carve((size_t)MTOK * HID * 2);
  _Float16* v16  = (_Float16*)carve((size_t)MTOK * HID * 2);
  _Float16* m16  = (_Float16*)carve((size_t)MTOK * MLPD * 2);
  _Float16* c16  = (_Float16*)carve((size_t)BATCH * HID * 2);
  float*    lg   = (float*)carve((size_t)BATCH * OUTD * 4);

  auto blks = [](long n) { return (unsigned)((n + 255) / 256); };

  // Weight conversion (f32 -> f16), qkv weights transposed to [k][n].
  cvt_kernel<<<blks((long)INDIM * HID), 256, 0, stream>>>(embed_W, wEmb, (long)INDIM * HID);
  cvt_qkv_kernel<<<blks((long)NL * NH * 4096), 256, 0, stream>>>(qW, wQ, NL * NH);
  cvt_qkv_kernel<<<blks((long)NL * NH * 4096), 256, 0, stream>>>(kW, wK, NL * NH);
  cvt_qkv_kernel<<<blks((long)NL * NH * 4096), 256, 0, stream>>>(vW, wV, NL * NH);
  cvt_kernel<<<blks((long)NL * HID * MLPD), 256, 0, stream>>>(mlp1_W, wM1, (long)NL * HID * MLPD);
  cvt_kernel<<<blks((long)NL * MLPD * HID), 256, 0, stream>>>(mlp2_W, wM2, (long)NL * MLPD * HID);
  cvt_kernel<<<blks((long)HID * OUTD), 256, 0, stream>>>(head_W, wHd, (long)HID * OUTD);

  pos_kernel<<<SEQ, HID, 0, stream>>>(pos);
  patchify_kernel<<<blks((long)MEMB * INDIM), 256, 0, stream>>>(x, Pm);
  clsrow_kernel<<<BATCH, HID, 0, stream>>>(cls, pos, h);

  // Patch embedding: h[b, 1+t, :] = P @ embed_W + b + pos
  gemm_wmma_kernel<EPI_EMBED><<<dim3(HID / 64, MEMB / 64, 1), 256, 0, stream>>>(
      Pm, INDIM, 0, wEmb, HID, 0, embed_b, 0,
      h, HID, nullptr, 0, 0, pos, MEMB, HID, INDIM);

  for (int l = 0; l < NL; ++l) {
    // LN1 -> y16
    ln_kernel<<<MTOK / 8, 256, 0, stream>>>(h, ln1_g + (long)l * HID, ln1_b + (long)l * HID,
                                            y16, MTOK);
    // Per-head Q/K/V projections (blockIdx.z = head)
    const long wOff = (long)l * NH * DH * DH;
    gemm_wmma_kernel<EPI_F16><<<dim3(1, MTOK / 64, NH), 256, 0, stream>>>(
        y16, HID, DH, wQ + wOff, DH, (long)DH * DH, qb + (long)l * NH * DH, DH,
        nullptr, 0, q16, HID, DH, nullptr, MTOK, DH, DH);
    gemm_wmma_kernel<EPI_F16><<<dim3(1, MTOK / 64, NH), 256, 0, stream>>>(
        y16, HID, DH, wK + wOff, DH, (long)DH * DH, kb + (long)l * NH * DH, DH,
        nullptr, 0, k16, HID, DH, nullptr, MTOK, DH, DH);
    gemm_wmma_kernel<EPI_F16><<<dim3(1, MTOK / 64, NH), 256, 0, stream>>>(
        y16, HID, DH, wV + wOff, DH, (long)DH * DH, vb + (long)l * NH * DH, DH,
        nullptr, 0, v16, HID, DH, nullptr, MTOK, DH, DH);
    // Attention, accumulated into h (residual)
    attn_kernel<<<BATCH * NH, 128, 0, stream>>>(q16, k16, v16, h);
    // LN2 -> y16
    ln_kernel<<<MTOK / 8, 256, 0, stream>>>(h, ln2_g + (long)l * HID, ln2_b + (long)l * HID,
                                            y16, MTOK);
    // MLP1 + exact GELU -> m16
    gemm_wmma_kernel<EPI_F16_GELU><<<dim3(MLPD / 64, MTOK / 64, 1), 256, 0, stream>>>(
        y16, HID, 0, wM1 + (long)l * HID * MLPD, MLPD, 0, mlp1_b + (long)l * MLPD, 0,
        nullptr, 0, m16, MLPD, 0, nullptr, MTOK, MLPD, HID);
    // MLP2, residual add into h
    gemm_wmma_kernel<EPI_F32_RES><<<dim3(HID / 64, MTOK / 64, 1), 256, 0, stream>>>(
        m16, MLPD, 0, wM2 + (long)l * MLPD * HID, HID, 0, mlp2_b + (long)l * HID, 0,
        h, HID, nullptr, 0, 0, nullptr, MTOK, HID, MLPD);
  }

  // Head on CLS token + softmax
  extract_cls_kernel<<<BATCH, HID, 0, stream>>>(h, c16);
  gemm_wmma_kernel<EPI_F32><<<dim3((OUTD + 63) / 64, BATCH / 64, 1), 256, 0, stream>>>(
      c16, HID, 0, wHd, OUTD, 0, head_b, 0,
      lg, OUTD, nullptr, 0, 0, nullptr, BATCH, OUTD, HID);
  softmax_kernel<<<BATCH, 256, 0, stream>>>(lg, (float*)d_out);
}
